// FRPN_47579647705425
// MI455X (gfx1250) — matrix operand order repaired and checked by reading, not compiled, not run on hardware
//
#include <hip/hip_runtime.h>
#include <hip/hip_bf16.h>
#include <math.h>

// ---------------------------------------------------------------------------
// CDNA5 (gfx1250, wave32) implicit-GEMM conv via V_WMMA_F32_16X16X32_F16.
// 256 threads = 8 waves; tile 128 Cout (M) x 128 spatial (N), K-chunk 32.
// Waves 4(M) x 2(N); each wave: 32x64 sub-tile => 2x4 v_wmma accumulators.
// Software pipeline: global->reg stage for chunk c+1, WMMA on chunk c from
// LDS, reg->LDS store into the ping-pong buffer, ONE barrier per chunk.
// ---------------------------------------------------------------------------

typedef __attribute__((ext_vector_type(16))) _Float16 v16h;
typedef __attribute__((ext_vector_type(8)))  float    v8f;

#define TM  128
#define TN  128
#define TK  32
#define LDT 40   // padded LDS stride in halves (80B = 20 banks, conflict-free)

__global__ __launch_bounds__(256)
void conv_igemm_wmma(const float* __restrict__ src, const float* __restrict__ wgt,
                     const float* __restrict__ bias, float* __restrict__ dst,
                     int Cin, int Cout, int H, int Wd, int logW,
                     int ks, int pad, int relu)
{
    __shared__ _Float16 As[2][TM * LDT];   // [m][k] weights (f32 -> f16)
    __shared__ _Float16 Bs[2][TN * LDT];   // [n][k] im2col  (f32 -> f16)

    const int tid  = threadIdx.x;
    const int lane = tid & 31;
    const int wv   = tid >> 5;       // 0..7
    const int wm   = wv >> 1;        // 0..3 : 32-row M slab
    const int wn   = wv & 1;         // 0..1 : 64-col N slab

    const int img   = blockIdx.z;
    const int nBase = blockIdx.x * TN;
    const int mBase = blockIdx.y * TM;
    const int HW  = H * Wd;
    const int ks2 = ks * ks;
    const int Kt  = Cin * ks2;

    // Each thread owns K-slot kFix (constant!) and rows rowBase+8j, j=0..15
    const int kFix    = tid & 31;
    const int rowBase = tid >> 5;

    const float* srcImg = src + (size_t)img * Cin * HW;

    v8f acc[2][4];
    {
        v8f z = {0.f,0.f,0.f,0.f,0.f,0.f,0.f,0.f};
        #pragma unroll
        for (int i = 0; i < 2; ++i)
            #pragma unroll
            for (int j = 0; j < 4; ++j) acc[i][j] = z;
    }

    // ---- staging helpers: global -> regs, regs -> LDS --------------------
    auto loadStage = [&](int k0, _Float16* aR, _Float16* bR) {
        const int kk  = k0 + kFix;
        const bool kOk = kk < Kt;
        // A: weights row-contiguous in K, OIHW flat = [Cout][Cin*ks*ks]
        {
            const float* wp = wgt + (size_t)(mBase + rowBase) * Kt + kk;
            #pragma unroll
            for (int j = 0; j < 16; ++j) {
                int co = mBase + rowBase + 8 * j;
                float v = (kOk && co < Cout) ? wp[(size_t)(8 * j) * Kt] : 0.f;
                aR[j] = (_Float16)v;
            }
        }
        // B: im2col. (cin,ky,kx) computed ONCE (kFix constant per thread);
        // oy/ox via shift/mask since W is a power of two for every layer.
        {
            int cin = 0, ky = 0, kx = 0;
            if (kOk) { cin = kk / ks2; int r = kk - cin * ks2; ky = r / ks; kx = r - ky * ks; }
            const float* sp = srcImg + (size_t)cin * HW;
            const int wmask = Wd - 1;
            #pragma unroll
            for (int j = 0; j < 16; ++j) {
                int pos = nBase + rowBase + 8 * j;
                float v = 0.f;
                if (kOk && pos < HW) {
                    int oy = pos >> logW, ox = pos & wmask;
                    int iy = oy + ky - pad, ix = ox + kx - pad;
                    if (iy >= 0 && iy < H && ix >= 0 && ix < Wd)
                        v = sp[iy * Wd + ix];
                }
                bR[j] = (_Float16)v;
            }
        }
    };
    auto storeStage = [&](int buf, const _Float16* aR, const _Float16* bR) {
        _Float16* ap = &As[buf][rowBase * LDT + kFix];
        _Float16* bp = &Bs[buf][rowBase * LDT + kFix];
        #pragma unroll
        for (int j = 0; j < 16; ++j) {
            ap[8 * j * LDT] = aR[j];
            bp[8 * j * LDT] = bR[j];
        }
    };

    // ---- pipelined K loop ------------------------------------------------
    const int nChunks = (Kt + TK - 1) / TK;
    _Float16 aR[16], bR[16];
    loadStage(0, aR, bR);
    storeStage(0, aR, bR);
    __syncthreads();

    // fragment index constants per ISA 16-bit layouts:
    // A (16x32): lanes 0-15: K={0..7,16..23}; lanes 16-31: K={8..15,24..31}
    // B (32x16): lanes 0-15: K={0..7,16..23}; lanes 16-31: K={8..15,24..31}; N/M = lane%16
    const int rsel = lane & 15;
    const int kbA  = (lane < 16) ? 0 : 8;
    const int kbB  = (lane < 16) ? 0 : 8;

    for (int c = 0; c < nChunks; ++c) {
        const int cur  = c & 1;
        const bool more = (c + 1) < nChunks;

        if (more) {
            if (c + 2 < nChunks) {   // global_prefetch_b8 for chunk c+2 weights
                int co = mBase + (tid & 127);
                if (co < Cout)
                    __builtin_prefetch(&wgt[(size_t)co * Kt + (c + 2) * TK], 0, 1);
            }
            loadStage((c + 1) * TK, aR, bR);   // global -> regs (latency hidden)
        }

        // WMMA on chunk c from LDS[cur]
        const _Float16* Ab = As[cur];
        const _Float16* Bb = Bs[cur];
        v16h a[2], b[4];
        #pragma unroll
        for (int t = 0; t < 2; ++t) {
            const _Float16* ap = &Ab[(wm * 32 + t * 16 + rsel) * LDT + kbA];
            #pragma unroll
            for (int i = 0; i < 8; ++i) { a[t][i] = ap[i]; a[t][8 + i] = ap[16 + i]; }
        }
        #pragma unroll
        for (int t = 0; t < 4; ++t) {
            const _Float16* bp = &Bb[(wn * 64 + t * 16 + rsel) * LDT + kbB];
            #pragma unroll
            for (int i = 0; i < 8; ++i) { b[t][i] = bp[i]; b[t][8 + i] = bp[16 + i]; }
        }
        #pragma unroll
        for (int tm = 0; tm < 2; ++tm)
            #pragma unroll
            for (int tn = 0; tn < 4; ++tn)
                acc[tm][tn] = __builtin_amdgcn_wmma_f32_16x16x32_f16(
                    false, a[tm], false, b[tn], (short)0, acc[tm][tn], false, false);

        if (more) storeStage(cur ^ 1, aR, bR);  // regs -> other LDS buffer
        __syncthreads();                         // single barrier per chunk
    }

    // ---- writeback: C layout VGPR v -> M=v(+8 upper half-wave), N=lane%16
    float* dstImg = dst + (size_t)img * Cout * HW;
    const int nlane = lane & 15;
    const int hi    = lane >> 4;
    #pragma unroll
    for (int tm = 0; tm < 2; ++tm)
        #pragma unroll
        for (int tn = 0; tn < 4; ++tn) {
            int pos = nBase + wn * 64 + tn * 16 + nlane;
            #pragma unroll
            for (int v = 0; v < 8; ++v) {
                int co = mBase + wm * 32 + tm * 16 + v + 8 * hi;
                if (co < Cout && pos < HW) {
                    float val = acc[tm][tn][v] + bias[co];
                    if (relu) val = fmaxf(val, 0.f);
                    dstImg[(size_t)co * HW + pos] = val;
                }
            }
        }
}

// ---------------------------------------------------------------------------
// 2x2 max pool, stride 2
// ---------------------------------------------------------------------------
__global__ __launch_bounds__(256)
void maxpool2(const float* __restrict__ src, float* __restrict__ dst,
              int C, int H, int Wd, int N)
{
    int H2 = H >> 1, W2 = Wd >> 1;
    long long total = (long long)N * C * H2 * W2;
    long long idx = (long long)blockIdx.x * 256 + threadIdx.x;
    if (idx >= total) return;
    int w2 = (int)(idx % W2);
    int h2 = (int)((idx / W2) % H2);
    int c  = (int)((idx / ((long long)W2 * H2)) % C);
    int n  = (int)(idx / ((long long)W2 * H2 * C));
    const float* p = src + (((size_t)n * C + c) * H + 2 * h2) * Wd + 2 * w2;
    float v0 = p[0], v1 = p[1], v2 = p[Wd], v3 = p[Wd + 1];
    dst[idx] = fmaxf(fmaxf(v0, v1), fmaxf(v2, v3));
}

// ---------------------------------------------------------------------------
// RPN post: softmax fg, locs transpose, anchors, box decode + clip
// hh=ww=16, 9 anchors, 4 images. p = (y*16+x)*9 + a
// ---------------------------------------------------------------------------
__global__ __launch_bounds__(256)
void rpn_post(const float* __restrict__ scoreMap, const float* __restrict__ locMap,
              float* __restrict__ fgOut, float* __restrict__ locsOut,
              float* __restrict__ anchOut, float* __restrict__ boxes)
{
    const int HH = 16, WW = 16, A = 9, NP = HH * WW * A; // 2304
    int idx = blockIdx.x * 256 + threadIdx.x;
    if (idx >= 4 * NP) return;
    int img = idx / NP;
    int p   = idx - img * NP;
    int y = p / (WW * A);
    int r = p - y * (WW * A);
    int x = r / A;
    int a = r - x * A;

    const int HWp = HH * WW;
    float s0 = scoreMap[((size_t)img * 18 + (a * 2 + 0)) * HWp + y * WW + x];
    float s1 = scoreMap[((size_t)img * 18 + (a * 2 + 1)) * HWp + y * WW + x];
    float fg = 1.0f / (1.0f + expf(s0 - s1));
    fgOut[(size_t)img * NP + p] = fg;

    float l[4];
    #pragma unroll
    for (int j = 0; j < 4; ++j) {
        l[j] = locMap[((size_t)img * 36 + (a * 4 + j)) * HWp + y * WW + x];
        locsOut[((size_t)img * NP + p) * 4 + j] = l[j];
    }

    const float ratios[3] = {0.5f, 1.0f, 2.0f};
    const float scales[3] = {1.0f, 2.0f, 4.0f};
    float ratio = ratios[a / 3], scale = scales[a % 3];
    float ah = 16.0f * scale * sqrtf(ratio);
    float aw = 16.0f * scale * sqrtf(1.0f / ratio);
    float sy = y * 32.0f, sx = x * 32.0f;
    float a0 = fminf(fmaxf(sy + 8.0f - ah * 0.5f, 0.f), 512.f);
    float a1 = fminf(fmaxf(sx + 8.0f - aw * 0.5f, 0.f), 512.f);
    float a2 = fminf(fmaxf(sy + 8.0f + ah * 0.5f, 0.f), 512.f);
    float a3 = fminf(fmaxf(sx + 8.0f + aw * 0.5f, 0.f), 512.f);
    if (img == 0) {
        anchOut[(size_t)p * 4 + 0] = a0;
        anchOut[(size_t)p * 4 + 1] = a1;
        anchOut[(size_t)p * 4 + 2] = a2;
        anchOut[(size_t)p * 4 + 3] = a3;
    }

    float py = a0, px = a1, ph = a2 - a0, pw = a3 - a1;
    float dx = l[0], dy = l[1], dw = l[2], dh = l[3];
    float gx = px + dx * pw;
    float gy = py + dy * ph;
    float gw = pw * expf(dw);
    float gh = ph * expf(dh);
    float* bp = boxes + ((size_t)img * NP + p) * 4;
    bp[0] = fminf(fmaxf(gy, 0.f), 512.f);
    bp[1] = fminf(fmaxf(gx, 0.f), 512.f);
    bp[2] = fminf(fmaxf(gy + gh, 0.f), 512.f);
    bp[3] = fminf(fmaxf(gx + gw, 0.f), 512.f);
}

// ---------------------------------------------------------------------------
// top-10 per image (2304 scores), stable ties (lower index wins), gather boxes
// ---------------------------------------------------------------------------
__global__ __launch_bounds__(256)
void topk10(const float* __restrict__ fg, const float* __restrict__ boxes,
            float* __restrict__ rois)
{
    __shared__ float sv[2304];
    __shared__ float rv[256];
    __shared__ int   ri[256];
    const int NP = 2304;
    int img = blockIdx.x, tid = threadIdx.x;
    for (int i = tid; i < NP; i += 256) sv[i] = fg[(size_t)img * NP + i];
    __syncthreads();
    for (int t = 0; t < 10; ++t) {
        float bv = -1e30f; int bi = 0x7fffffff;
        for (int i = tid; i < NP; i += 256) {
            float v = sv[i];
            if (v > bv || (v == bv && i < bi)) { bv = v; bi = i; }
        }
        rv[tid] = bv; ri[tid] = bi;
        __syncthreads();
        for (int s = 128; s > 0; s >>= 1) {
            if (tid < s) {
                if (rv[tid + s] > rv[tid] ||
                    (rv[tid + s] == rv[tid] && ri[tid + s] < ri[tid])) {
                    rv[tid] = rv[tid + s]; ri[tid] = ri[tid + s];
                }
            }
            __syncthreads();
        }
        if (tid == 0) {
            int best = ri[0];
            #pragma unroll
            for (int j = 0; j < 4; ++j)
                rois[((size_t)img * 10 + t) * 4 + j] =
                    boxes[((size_t)img * NP + best) * 4 + j];
            sv[best] = -1e30f;
        }
        __syncthreads();
    }
}

// ---------------------------------------------------------------------------
// Host-side orchestration (all launches on `stream`; ws used as ping-pong)
// ---------------------------------------------------------------------------
extern "C" void kernel_launch(void* const* d_in, const int* in_sizes, int n_in,
                              void* d_out, int out_size, void* d_ws, size_t ws_size,
                              hipStream_t stream)
{
    (void)in_sizes; (void)n_in; (void)out_size; (void)ws_size;

    const float* x = (const float*)d_in[0];
    const float* rpnW = (const float*)d_in[27]; const float* rpnB = (const float*)d_in[28];
    const float* scW  = (const float*)d_in[29]; const float* scB  = (const float*)d_in[30];
    const float* lcW  = (const float*)d_in[31]; const float* lcB  = (const float*)d_in[32];

    char* ws = (char*)d_ws;
    const size_t HALF = (size_t)268435456; // 256 MiB: 4*64*512*512 f32
    float* bufA = (float*)ws;
    float* bufB = (float*)(ws + HALF);
    float* scoreOut = (float*)(ws + (size_t)8  * 1024 * 1024);
    float* locOut   = (float*)(ws + (size_t)12 * 1024 * 1024);
    float* boxes    = (float*)(ws + (size_t)16 * 1024 * 1024);

    auto conv = [&](const float* s, const float* w, const float* b, float* d,
                    int Cin, int Cout, int H, int Wd, int ks, int pad, int relu) {
        int logW = 31 - __builtin_clz((unsigned)Wd);
        dim3 grid((H * Wd + TN - 1) / TN, (Cout + TM - 1) / TM, 4);
        conv_igemm_wmma<<<grid, 256, 0, stream>>>(s, w, b, d, Cin, Cout, H, Wd,
                                                  logW, ks, pad, relu);
    };
    auto pool = [&](const float* s, float* d, int C, int H, int Wd) {
        long long total = 4LL * C * (H / 2) * (Wd / 2);
        maxpool2<<<(unsigned)((total + 255) / 256), 256, 0, stream>>>(s, d, C, H, Wd, 4);
    };
    auto bw = [&](int i) { return (const float*)d_in[1 + 2 * i]; };
    auto bb = [&](int i) { return (const float*)d_in[2 + 2 * i]; };

    // VGG16 backbone
    conv(x,    bw(0),  bb(0),  bufA, 3,   64,  512, 512, 3, 1, 1);
    conv(bufA, bw(1),  bb(1),  bufB, 64,  64,  512, 512, 3, 1, 1);
    pool(bufB, bufA, 64, 512, 512);                       // -> 256
    conv(bufA, bw(2),  bb(2),  bufB, 64,  128, 256, 256, 3, 1, 1);
    conv(bufB, bw(3),  bb(3),  bufA, 128, 128, 256, 256, 3, 1, 1);
    pool(bufA, bufB, 128, 256, 256);                      // -> 128
    conv(bufB, bw(4),  bb(4),  bufA, 128, 256, 128, 128, 3, 1, 1);
    conv(bufA, bw(5),  bb(5),  bufB, 256, 256, 128, 128, 3, 1, 1);
    conv(bufB, bw(6),  bb(6),  bufA, 256, 256, 128, 128, 3, 1, 1);
    pool(bufA, bufB, 256, 128, 128);                      // -> 64
    conv(bufB, bw(7),  bb(7),  bufA, 256, 512, 64, 64, 3, 1, 1);
    conv(bufA, bw(8),  bb(8),  bufB, 512, 512, 64, 64, 3, 1, 1);
    conv(bufB, bw(9),  bb(9),  bufA, 512, 512, 64, 64, 3, 1, 1);
    pool(bufA, bufB, 512, 64, 64);                        // -> 32
    conv(bufB, bw(10), bb(10), bufA, 512, 512, 32, 32, 3, 1, 1);
    conv(bufA, bw(11), bb(11), bufB, 512, 512, 32, 32, 3, 1, 1);
    conv(bufB, bw(12), bb(12), bufA, 512, 512, 32, 32, 3, 1, 1);
    pool(bufA, bufB, 512, 32, 32);                        // feats in bufB (16x16)

    // RPN head
    conv(bufB, rpnW, rpnB, bufA, 512, 512, 16, 16, 3, 1, 1);
    conv(bufA, scW,  scB,  scoreOut, 512, 18, 16, 16, 1, 0, 0);
    conv(bufA, lcW,  lcB,  locOut,   512, 36, 16, 16, 1, 0, 0);

    // Outputs: fg | rpn_locs | anchors | rois (flat, in return order)
    float* o       = (float*)d_out;
    float* out_fg  = o;                       // 4*2304
    float* out_loc = o + 4 * 2304;            // 4*2304*4
    float* out_anc = out_loc + 4 * 2304 * 4;  // 2304*4
    float* out_roi = out_anc + 2304 * 4;      // 4*10*4

    rpn_post<<<(4 * 2304 + 255) / 256, 256, 0, stream>>>(
        scoreOut, locOut, out_fg, out_loc, out_anc, boxes);
    topk10<<<4, 256, 0, stream>>>(out_fg, boxes, out_roi);
}
